// MultiHeadCrossAttention_58798102282886
// MI455X (gfx1250) — matrix operand order repaired and checked by reading, not compiled, not run on hardware
//
#include <hip/hip_runtime.h>

// ---------------------------------------------------------------------------
// MI455X (gfx1250) multi-head cross attention with relative position bias.
// wave32, WMMA bf16 16x16x32 with f32 accumulation everywhere.
// ---------------------------------------------------------------------------

#define DEVFN __device__ __forceinline__

typedef __attribute__((ext_vector_type(16))) __bf16 v16bf;
typedef __attribute__((ext_vector_type(8)))  __bf16 v8bf;
typedef __attribute__((ext_vector_type(8)))  float  v8f;
typedef __attribute__((ext_vector_type(4)))  float  v4f;

static constexpr int NFEAT = 512;
static constexpr int NHEAD = 8;
static constexpr int DK    = 64;
static constexpr int BATCH = 8;
static constexpr int T     = 1024;
static constexpr int MTOK  = BATCH * T;   // 8192 token rows

DEVFN v8f wmma_bf16(v16bf a, v16bf b, v8f c) {
  // D = A(16x32 bf16) * B(32x16 bf16) + C(16x16 f32)
  return __builtin_amdgcn_wmma_f32_16x16x32_bf16(false, a, false, b, (short)0, c,
                                                 false, false);
}

DEVFN v16bf cat8(v8bf lo, v8bf hi) {
  return __builtin_shufflevector(lo, hi, 0, 1, 2, 3, 4, 5, 6, 7,
                                 8, 9, 10, 11, 12, 13, 14, 15);
}

// A-fragment (16x32 bf16) for this lane from a row-major row pointer.
// Lane holds row m = lane%16; VGPRs 0-3 = K[hi*16 .. hi*16+7], VGPRs 4-7 =
// K[hi*16+8 .. hi*16+15]  (hi = lane/16). Two contiguous 16B loads.
DEVFN v16bf loadA16(const __bf16* row, int k0, int hi) {
  v8bf lo = *(const v8bf*)(row + k0 + hi * 16);
  v8bf hi8 = *(const v8bf*)(row + k0 + hi * 16 + 8);
  return cat8(lo, hi8);
}

// Convert 16 contiguous f32 to a bf16 B-fragment half (used for pos_k on the fly).
DEVFN v16bf cvt16(const float* p) {
  v4f f0 = ((const v4f*)p)[0];
  v4f f1 = ((const v4f*)p)[1];
  v4f f2 = ((const v4f*)p)[2];
  v4f f3 = ((const v4f*)p)[3];
  v16bf r;
#pragma unroll
  for (int i = 0; i < 4; ++i) {
    r[i]      = (__bf16)f0[i];
    r[4 + i]  = (__bf16)f1[i];
    r[8 + i]  = (__bf16)f2[i];
    r[12 + i] = (__bf16)f3[i];
  }
  return r;
}

// ---------------------------------------------------------------------------
// LayerNorm (row of 512 f32) -> bf16.  One row per 128-thread block.
// ---------------------------------------------------------------------------
__global__ __launch_bounds__(128) void ln_bf16_kernel(
    const float* __restrict__ x, const float* __restrict__ g,
    const float* __restrict__ bta, __bf16* __restrict__ out) {
  const int row = blockIdx.x;
  const int tid = threadIdx.x;
  const float* xr = x + (size_t)row * NFEAT;
  float4 v = ((const float4*)xr)[tid];
  float s  = v.x + v.y + v.z + v.w;
  float s2 = v.x * v.x + v.y * v.y + v.z * v.z + v.w * v.w;
#pragma unroll
  for (int m = 16; m >= 1; m >>= 1) {
    s  += __shfl_xor(s, m, 32);
    s2 += __shfl_xor(s2, m, 32);
  }
  __shared__ float red[4][2];
  const int wave = tid >> 5, lane = tid & 31;
  if (lane == 0) { red[wave][0] = s; red[wave][1] = s2; }
  __syncthreads();
  s  = red[0][0] + red[1][0] + red[2][0] + red[3][0];
  s2 = red[0][1] + red[1][1] + red[2][1] + red[3][1];
  const float mu  = s * (1.0f / NFEAT);
  const float var = s2 * (1.0f / NFEAT) - mu * mu;
  const float inv = rsqrtf(var + 1e-5f);
  const int c = tid * 4;
  __bf16* orow = out + (size_t)row * NFEAT + c;
  orow[0] = (__bf16)((v.x - mu) * inv * g[c + 0] + bta[c + 0]);
  orow[1] = (__bf16)((v.y - mu) * inv * g[c + 1] + bta[c + 1]);
  orow[2] = (__bf16)((v.z - mu) * inv * g[c + 2] + bta[c + 2]);
  orow[3] = (__bf16)((v.w - mu) * inv * g[c + 3] + bta[c + 3]);
}

// ---------------------------------------------------------------------------
// f32 -> bf16 elementwise (weights).
// ---------------------------------------------------------------------------
__global__ __launch_bounds__(256) void f32_to_bf16_kernel(
    const float* __restrict__ in, __bf16* __restrict__ out, int n) {
  const int i = blockIdx.x * 256 + threadIdx.x;
  if (i < n) out[i] = (__bf16)in[i];
}

// ---------------------------------------------------------------------------
// GEMM: out = A(M,512) @ W(512,512)^T + bias.  A, W bf16; acc f32.
// W is (f, c) row-major so B-fragment columns (16 consecutive c for feature f)
// are contiguous 32B loads — no transpose needed.
// Block = 4 waves; wave computes 16(M) x 64(N); block = 64x64.
// mode 0: bf16 row-major (token, f)          [q_proj, k_proj]
// mode 1: bf16 transposed per head (b,h,d,s) [v_proj -> vT, via LDS transpose]
// mode 2: f32 row-major, (acc+bias)*scale    [final output]
// ---------------------------------------------------------------------------
__global__ __launch_bounds__(128) void gemm_nt_kernel(
    const __bf16* __restrict__ A, const __bf16* __restrict__ W,
    const float* __restrict__ bias, const float* __restrict__ scale,
    void* __restrict__ outp, int mode) {
  const int wave = threadIdx.x >> 5, lane = threadIdx.x & 31;
  const int mrow = lane & 15, hi = lane >> 4;
  const int m0 = blockIdx.x * 64 + wave * 16;
  const int n0 = blockIdx.y * 64;
  const __bf16* Arow = A + (size_t)(m0 + mrow) * NFEAT;

  v8f acc[4] = {};
  for (int k0 = 0; k0 < NFEAT; k0 += 32) {
    v16bf a = loadA16(Arow, k0, hi);
#pragma unroll
    for (int nt = 0; nt < 4; ++nt) {
      const int f = n0 + nt * 16 + mrow;  // lane n-index
      v16bf b = *(const v16bf*)(W + (size_t)f * NFEAT + k0 + hi * 16);
      acc[nt] = wmma_bf16(a, b, acc[nt]);
    }
  }

  if (mode == 0) {
    __bf16* out = (__bf16*)outp;
#pragma unroll
    for (int nt = 0; nt < 4; ++nt) {
      const int f = n0 + nt * 16 + mrow;
      const float bv = bias[f];
#pragma unroll
      for (int r = 0; r < 8; ++r) {
        const int row = m0 + r + 8 * hi;  // D: lanes 0-15 -> M=r, 16-31 -> M=8+r
        out[(size_t)row * NFEAT + f] = (__bf16)(acc[nt][r] + bv);
      }
    }
  } else if (mode == 2) {
    float* out = (float*)outp;
#pragma unroll
    for (int nt = 0; nt < 4; ++nt) {
      const int f = n0 + nt * 16 + mrow;
      const float bv = bias[f];
      const float ls = scale[f];
#pragma unroll
      for (int r = 0; r < 8; ++r) {
        const int row = m0 + r + 8 * hi;
        out[(size_t)row * NFEAT + f] = (acc[nt][r] + bv) * ls;
      }
    }
  } else {
    // mode 1: per-wave LDS transpose, then contiguous 32B row stores to (b,h,d,s)
    __shared__ __bf16 tr[4][64 * 24];  // stride 24 elems = 48B (16B aligned)
    __bf16* L = tr[wave];
#pragma unroll
    for (int nt = 0; nt < 4; ++nt) {
      const int fl = nt * 16 + mrow;
      const float bv = bias[n0 + fl];
#pragma unroll
      for (int r = 0; r < 8; ++r)
        L[fl * 24 + r + 8 * hi] = (__bf16)(acc[nt][r] + bv);
    }
    asm volatile("s_wait_dscnt 0" ::: "memory");  // wave-private region: no barrier
    const int b = m0 / T, sbase = m0 % T;
    __bf16* out = (__bf16*)outp;
#pragma unroll
    for (int i = 0; i < 2; ++i) {
      const int fl = lane + 32 * i;
      const int f = n0 + fl;
      const int d = f & (DK - 1), h = f >> 6;
      v8bf r0 = *(const v8bf*)(L + fl * 24);
      v8bf r1 = *(const v8bf*)(L + fl * 24 + 8);
      __bf16* dst = out + (((size_t)b * NHEAD + h) * DK + d) * T + sbase;
      *(v8bf*)dst = r0;
      *(v8bf*)(dst + 8) = r1;
    }
  }
}

// ---------------------------------------------------------------------------
// Relative-position scores: Bp[b,h,t,s] = sum_d q[b,h,t,d] * pos_k[t,s,d].
// Per-t GEMM: M = 64 (b*h), K = 64 (d), N = s.  pos_k converted f32->bf16
// on the fly (B-columns are contiguous in memory).
// Grid: (t, s/64); block = 4 waves, wave = 16(bh) x 64(s).
// ---------------------------------------------------------------------------
__global__ __launch_bounds__(128) void bp_gemm_kernel(
    const __bf16* __restrict__ qp, const float* __restrict__ pos_k,
    __bf16* __restrict__ Bp) {
  const int t = blockIdx.x;
  const int s0 = blockIdx.y * 64;
  const int wave = threadIdx.x >> 5, lane = threadIdx.x & 31;
  const int mrow = lane & 15, hi = lane >> 4;
  const int bh = wave * 16 + mrow;
  const int b = bh >> 3, h = bh & 7;
  const __bf16* Arow = qp + (size_t)(b * T + t) * NFEAT + h * DK;
  v16bf a0 = loadA16(Arow, 0, hi);
  v16bf a1 = loadA16(Arow, 32, hi);
#pragma unroll
  for (int nt = 0; nt < 4; ++nt) {
    const int s = s0 + nt * 16 + mrow;
    const float* pk = pos_k + ((size_t)t * T + s) * DK;
    v16bf b0 = cvt16(pk + hi * 16);
    v16bf b1 = cvt16(pk + 32 + hi * 16);
    v8f acc = {};
    acc = wmma_bf16(a0, b0, acc);
    acc = wmma_bf16(a1, b1, acc);
#pragma unroll
    for (int r = 0; r < 8; ++r) {
      const int rbh = wave * 16 + r + 8 * hi;
      Bp[((size_t)rbh * T + t) * T + s] = (__bf16)acc[r];
    }
  }
}

// ---------------------------------------------------------------------------
// Flash attention.  Grid: (b*h, T1/64); block = 4 waves; each wave owns its
// own 16 query rows (no cross-wave reductions).  Per 32-wide s step:
//   4 WMMA for scores, online softmax (16-lane shfl reductions),
//   P staged bf16 through wave-private LDS -> A-fragment, 4 WMMA for P@V.
// ---------------------------------------------------------------------------
__global__ __launch_bounds__(128) void flash_attn_kernel(
    const __bf16* __restrict__ qp, const __bf16* __restrict__ kp,
    const __bf16* __restrict__ vt, const __bf16* __restrict__ Bp,
    const int* __restrict__ mask, __bf16* __restrict__ xout) {
  const int bh = blockIdx.x;
  const int b = bh >> 3, h = bh & 7;
  const int wave = threadIdx.x >> 5, lane = threadIdx.x & 31;
  const int mrow = lane & 15, hi = lane >> 4;
  const int t0 = blockIdx.y * 64 + wave * 16;

  __shared__ __bf16 plds[4][16 * 56];  // stride 56 elems: 16B aligned, no bank conflict
  __bf16* P = plds[wave];

  const __bf16* qrow = qp + (size_t)(b * T + t0 + mrow) * NFEAT + h * DK;
  const v16bf aq0 = loadA16(qrow, 0, hi);
  const v16bf aq1 = loadA16(qrow, 32, hi);

  const __bf16* kbase = kp + (size_t)b * T * NFEAT + h * DK;
  const __bf16* vbase = vt + (size_t)bh * DK * T;     // (d, s) layout
  const __bf16* bpbase = Bp + (size_t)bh * T * T;     // (t, s) for this bh
  const int* mbase = mask + (size_t)b * T * T;

  v8f o[4] = {};
  float mrun[8], lrun[8];
#pragma unroll
  for (int r = 0; r < 8; ++r) { mrun[r] = -1e30f; lrun[r] = 0.f; }

  for (int s0 = 0; s0 < T; s0 += 32) {
    float sc[2][8];
#pragma unroll
    for (int j = 0; j < 2; ++j) {
      const int scol = s0 + j * 16 + mrow;  // lane n-index -> key position
      const __bf16* krow = kbase + (size_t)scol * NFEAT;
      v16bf bk0 = *(const v16bf*)(krow + hi * 16);
      v16bf bk1 = *(const v16bf*)(krow + 32 + hi * 16);
      v8f c = {};
      c = wmma_bf16(aq0, bk0, c);
      c = wmma_bf16(aq1, bk1, c);
#pragma unroll
      for (int r = 0; r < 8; ++r) {
        const int trow = t0 + r + 8 * hi;
        const float bpv = (float)bpbase[(size_t)trow * T + scol];
        const int mv = mbase[(size_t)trow * T + scol];
        sc[j][r] = mv ? (c[r] + bpv) * 0.125f : -1e30f;  // 1/sqrt(64)
      }
    }
    // Online softmax per row (rows live across 16 lanes of each half-wave).
#pragma unroll
    for (int r = 0; r < 8; ++r) {
      float mx = fmaxf(sc[0][r], sc[1][r]);
#pragma unroll
      for (int d = 8; d >= 1; d >>= 1) mx = fmaxf(mx, __shfl_xor(mx, d, 32));
      const float mnew = fmaxf(mrun[r], mx);
      float p0 = __expf(sc[0][r] - mnew);
      float p1 = __expf(sc[1][r] - mnew);
      if (sc[0][r] <= -1e29f) p0 = 0.f;
      if (sc[1][r] <= -1e29f) p1 = 0.f;
      float rs = p0 + p1;
#pragma unroll
      for (int d = 8; d >= 1; d >>= 1) rs += __shfl_xor(rs, d, 32);
      const float al = __expf(mrun[r] - mnew);
      lrun[r] = lrun[r] * al + rs;
      mrun[r] = mnew;
      sc[0][r] = p0;
      sc[1][r] = p1;
#pragma unroll
      for (int nt = 0; nt < 4; ++nt) o[nt][r] *= al;
    }
    // Stage P (16x32 bf16) into wave-private LDS, re-read as A-fragment.
#pragma unroll
    for (int j = 0; j < 2; ++j)
#pragma unroll
      for (int r = 0; r < 8; ++r)
        P[(r + 8 * hi) * 56 + j * 16 + mrow] = (__bf16)sc[j][r];
    asm volatile("s_wait_dscnt 0" ::: "memory");
    v16bf ap = cat8(*(const v8bf*)(P + mrow * 56 + hi * 16),
                    *(const v8bf*)(P + mrow * 56 + hi * 16 + 8));
#pragma unroll
    for (int nt = 0; nt < 4; ++nt) {
      const __bf16* vrow = vbase + (size_t)(nt * 16 + mrow) * T + s0 + hi * 16;
      v16bf bv = *(const v16bf*)vrow;
      o[nt] = wmma_bf16(ap, bv, o[nt]);
    }
  }
  // Epilogue: normalize and store x (b, t, h*64+d) bf16.
#pragma unroll
  for (int r = 0; r < 8; ++r) {
    const float invl = lrun[r] > 0.f ? 1.0f / lrun[r] : 0.f;
    const int trow = t0 + r + 8 * hi;
#pragma unroll
    for (int nt = 0; nt < 4; ++nt)
      xout[(size_t)(b * T + trow) * NFEAT + h * DK + nt * 16 + mrow] =
          (__bf16)(o[nt][r] * invl);
  }
}

// ---------------------------------------------------------------------------
extern "C" void kernel_launch(void* const* d_in, const int* in_sizes, int n_in,
                              void* d_out, int out_size, void* d_ws,
                              size_t ws_size, hipStream_t stream) {
  (void)in_sizes; (void)n_in; (void)out_size; (void)ws_size;
  const float* q      = (const float*)d_in[0];
  const float* k      = (const float*)d_in[1];
  const float* v      = (const float*)d_in[2];
  const float* pos_k  = (const float*)d_in[3];
  const int*   mask   = (const int*)d_in[4];
  const float* ln_q_g = (const float*)d_in[5];
  const float* ln_q_b = (const float*)d_in[6];
  const float* ln_kv_g= (const float*)d_in[7];
  const float* ln_kv_b= (const float*)d_in[8];
  const float* Wq = (const float*)d_in[9];
  const float* bq = (const float*)d_in[10];
  const float* Wk = (const float*)d_in[11];
  const float* bk = (const float*)d_in[12];
  const float* Wv = (const float*)d_in[13];
  const float* bv = (const float*)d_in[14];
  const float* Wo = (const float*)d_in[15];
  const float* bo = (const float*)d_in[16];
  const float* lscale = (const float*)d_in[17];
  float* out = (float*)d_out;

  char* ws = (char*)d_ws;
  size_t off = 0;
  auto take = [&](size_t bytes) -> void* {
    void* p = ws + off;
    off += (bytes + 255) & ~(size_t)255;
    return p;
  };
  const size_t tokb = (size_t)MTOK * NFEAT * sizeof(__bf16);  // 8 MB each
  const size_t wb   = (size_t)NFEAT * NFEAT * sizeof(__bf16); // 512 KB each
  __bf16* qn    = (__bf16*)take(tokb);
  __bf16* kn    = (__bf16*)take(tokb);
  __bf16* vn    = (__bf16*)take(tokb);
  __bf16* wqb   = (__bf16*)take(wb);
  __bf16* wkb   = (__bf16*)take(wb);
  __bf16* wvb   = (__bf16*)take(wb);
  __bf16* wob   = (__bf16*)take(wb);
  __bf16* qproj = (__bf16*)take(tokb);
  __bf16* kproj = (__bf16*)take(tokb);
  __bf16* vtb   = (__bf16*)take(tokb);
  __bf16* xbuf  = (__bf16*)take(tokb);
  __bf16* bpbuf = (__bf16*)take((size_t)BATCH * NHEAD * T * T * sizeof(__bf16)); // 128 MB

  // 1) LayerNorm + bf16
  ln_bf16_kernel<<<MTOK, 128, 0, stream>>>(q, ln_q_g, ln_q_b, qn);
  ln_bf16_kernel<<<MTOK, 128, 0, stream>>>(k, ln_kv_g, ln_kv_b, kn);
  ln_bf16_kernel<<<MTOK, 128, 0, stream>>>(v, ln_kv_g, ln_kv_b, vn);

  // 2) Weight conversion
  const int WN = NFEAT * NFEAT;
  f32_to_bf16_kernel<<<WN / 256, 256, 0, stream>>>(Wq, wqb, WN);
  f32_to_bf16_kernel<<<WN / 256, 256, 0, stream>>>(Wk, wkb, WN);
  f32_to_bf16_kernel<<<WN / 256, 256, 0, stream>>>(Wv, wvb, WN);
  f32_to_bf16_kernel<<<WN / 256, 256, 0, stream>>>(Wo, wob, WN);

  // 3) Projections (WMMA)
  dim3 gproj(MTOK / 64, NFEAT / 64);
  gemm_nt_kernel<<<gproj, 128, 0, stream>>>(qn, wqb, bq, nullptr, qproj, 0);
  gemm_nt_kernel<<<gproj, 128, 0, stream>>>(kn, wkb, bk, nullptr, kproj, 0);
  gemm_nt_kernel<<<gproj, 128, 0, stream>>>(vn, wvb, bv, nullptr, vtb, 1);

  // 4) Relative-position scores (WMMA, pos_k converted on the fly)
  bp_gemm_kernel<<<dim3(T, T / 64), 128, 0, stream>>>(qproj, pos_k, bpbuf);

  // 5) Flash attention (WMMA)
  flash_attn_kernel<<<dim3(BATCH * NHEAD, T / 64), 128, 0, stream>>>(
      qproj, kproj, vtb, bpbuf, mask, xbuf);

  // 6) Output projection + layer_scale (WMMA, fp32 out)
  gemm_nt_kernel<<<gproj, 128, 0, stream>>>(xbuf, wob, bo, lscale, out, 2);
}
